// Cfc_300647710890
// MI455X (gfx1250) — matrix operand order repaired and checked by reading, not compiled, not run on hardware
//
#include <hip/hip_runtime.h>

#define B_   512
#define T_   256
#define F_   128
#define H_   256
#define U_   512
#define OUT_ 128
#define FH_  384   // F_ + H_

#define ZSTR 392   // 384 + 8 pad (halfs) -> 784B rows, 16B aligned, bank-spread
#define USTR 520   // 512 + 8 pad (halfs) -> 1040B rows

typedef __attribute__((ext_vector_type(16))) _Float16 v16h;
typedef __attribute__((ext_vector_type(8)))  _Float16 v8h;
typedef __attribute__((ext_vector_type(8)))  float    v8f;

__device__ __forceinline__ v8f zero8() {
  v8f z = {0.f, 0.f, 0.f, 0.f, 0.f, 0.f, 0.f, 0.f};
  return z;
}

__device__ __forceinline__ float fast_tanh(float x) {
  // tanh(x) = 1 - 2/(exp(2x)+1), exp via v_exp_f32 (exp2)
  float e2 = __builtin_amdgcn_exp2f(x * 2.8853900817779268f); // 2*log2(e)
  return 1.0f - 2.0f * __builtin_amdgcn_rcpf(e2 + 1.0f);
}
__device__ __forceinline__ float lecun_act(float x) { return 1.7159f * fast_tanh(0.666f * x); }
__device__ __forceinline__ float fast_sigmoid(float x) {
  return __builtin_amdgcn_rcpf(1.0f + __builtin_amdgcn_exp2f(-1.4426950408889634f * x));
}

__device__ __forceinline__ v8f wmma16(v16h a, v16h b, v8f c) {
  return __builtin_amdgcn_wmma_f32_16x16x32_f16(false, a, false, b, (short)0, c, false, false);
}

// A-fragment (16x32 f16, M=row) from LDS, row-major with `stride` halfs.
// Layout used here (contiguous K, not ISA table order): lanes 0-15 row M=lane, elems0..15 = K k0..k0+15;
//             lanes 16-31 same rows, K offset +16.
__device__ __forceinline__ v16h load_a_frag(const _Float16* s, int stride, int k0, int lane) {
  const int row = lane & 15;
  const int sel = lane >> 4;
  const _Float16* p = s + row * stride + k0 + sel * 16;
  v8h lo = *(const v8h*)p;        // K = k0 + sel*16 + 0..7
  v8h hi = *(const v8h*)(p + 8);  // K = k0 + sel*16 + 8..15
  v16h r;
#pragma unroll
  for (int i = 0; i < 8; ++i) { r[i] = lo[i]; r[i + 8] = hi[i]; }
  return r;
}

// B-fragment (32x16 f16, N=col) from transposed weights Wt[N][K] row-major (strideK halfs).
// Layout used here: lane = column (lanes 16-31: K offset +16); elems = contiguous K -> one 32B load.
__device__ __forceinline__ v16h load_b_frag(const _Float16* wt, int strideK, int n0, int k0, int lane) {
  const int col = lane & 15;
  const int sel = lane >> 4;
  return *(const v16h*)(wt + (size_t)(n0 + col) * strideK + (k0 + sel * 16));
}

// Transpose fp32 [K][N] row-major -> f16 [N][K] row-major.
__global__ void cfc_w_transpose(const float* __restrict__ src, _Float16* __restrict__ dst,
                                int K, int N) {
  int idx = blockIdx.x * blockDim.x + threadIdx.x;
  if (idx >= K * N) return;
  int n = idx / K;
  int k = idx - n * K;
  dst[(size_t)n * K + k] = (_Float16)src[(size_t)k * N + n];
}

__global__ __launch_bounds__(256, 1) void cfc_main(
    const float* __restrict__ x, const float* __restrict__ ts,
    const _Float16* __restrict__ W1t, const float* __restrict__ b1,
    const _Float16* __restrict__ W2t, const float* __restrict__ b2,
    const _Float16* __restrict__ Wf1t, const float* __restrict__ bf1,
    const _Float16* __restrict__ Wf2t, const float* __restrict__ bf2,
    const _Float16* __restrict__ Wtat, const float* __restrict__ bta,
    const _Float16* __restrict__ Wtbt, const float* __restrict__ btb,
    const _Float16* __restrict__ Wot, const float* __restrict__ bo,
    float* __restrict__ out) {
  __shared__ alignas(32) _Float16 z_s[16 * ZSTR];   // [x(0:128) | h(128:384)] f16
  __shared__ alignas(32) _Float16 z1_s[16 * USTR];  // layer-1 activations
  __shared__ alignas(32) _Float16 z2_s[16 * USTR];  // layer-2 activations
  __shared__ float t_s[16];

  const int tid = threadIdx.x;
  const int lane = tid & 31;
  const int wave = tid >> 5;
  const int mb = blockIdx.x; // batch tile: rows mb*16 .. mb*16+15
  const int lcol = lane & 15;
  const int rowbase = (lane >> 4) * 8;

  // h0 = 0
  for (int i = tid; i < 16 * H_; i += 256) {
    int m = i >> 8, c = i & (H_ - 1);
    z_s[m * ZSTR + F_ + c] = (_Float16)0.0f;
  }

#pragma unroll 1
  for (int step = 0; step < T_; ++step) {
    // ---- S0: stage x_t tile (f32->f16) and t scalars into LDS ----
    {
      int m = tid >> 4, c0 = (tid & 15) * 8;
      const float* xp = x + ((size_t)(mb * 16 + m) * T_ + step) * F_ + c0;
      _Float16* zp = z_s + m * ZSTR + c0;
#pragma unroll
      for (int i = 0; i < 8; ++i) zp[i] = (_Float16)xp[i];
      if (tid < 16) t_s[tid] = ts[(size_t)(mb * 16 + tid) * T_ + step] * (1.0f / 256.0f);
    }
    __syncthreads();

    // ---- S1: z1 = lecun(z @ W1 + b1)   [16x384]x[384x512] ----
    {
      v8f acc[4] = {zero8(), zero8(), zero8(), zero8()};
      const int n0 = wave * 64;
      for (int k0 = 0; k0 < FH_; k0 += 32) {
        v16h a = load_a_frag(z_s, ZSTR, k0, lane);
#pragma unroll
        for (int i = 0; i < 4; ++i) {
          v16h b = load_b_frag(W1t, FH_, n0 + i * 16, k0, lane);
          acc[i] = wmma16(a, b, acc[i]);
        }
      }
#pragma unroll
      for (int i = 0; i < 4; ++i) {
        const int col = n0 + i * 16 + lcol;
        const float bias = b1[col];
#pragma unroll
        for (int r = 0; r < 8; ++r)
          z1_s[(rowbase + r) * USTR + col] = (_Float16)lecun_act(acc[i][r] + bias);
      }
    }
    __syncthreads();

    // ---- S2: z2 = lecun(z1 @ W2 + b2)  [16x512]x[512x512] ----
    {
      v8f acc[4] = {zero8(), zero8(), zero8(), zero8()};
      const int n0 = wave * 64;
      for (int k0 = 0; k0 < U_; k0 += 32) {
        v16h a = load_a_frag(z1_s, USTR, k0, lane);
#pragma unroll
        for (int i = 0; i < 4; ++i) {
          v16h b = load_b_frag(W2t, U_, n0 + i * 16, k0, lane);
          acc[i] = wmma16(a, b, acc[i]);
        }
      }
#pragma unroll
      for (int i = 0; i < 4; ++i) {
        const int col = n0 + i * 16 + lcol;
        const float bias = b2[col];
#pragma unroll
        for (int r = 0; r < 8; ++r)
          z2_s[(rowbase + r) * USTR + col] = (_Float16)lecun_act(acc[i][r] + bias);
      }
    }
    __syncthreads();

    // ---- S3: four heads + gate -> h_new, written into z_s h-columns ----
    {
      v8f aF1[2] = {zero8(), zero8()}, aF2[2] = {zero8(), zero8()};
      v8f aTA[2] = {zero8(), zero8()}, aTB[2] = {zero8(), zero8()};
      const int j0 = wave * 32; // each wave owns 32 h-columns (2 tiles) across all 4 heads
      for (int k0 = 0; k0 < U_; k0 += 32) {
        v16h a = load_a_frag(z2_s, USTR, k0, lane);
#pragma unroll
        for (int jt = 0; jt < 2; ++jt) {
          const int n0 = j0 + jt * 16;
          v16h b;
          b = load_b_frag(Wf1t, U_, n0, k0, lane); aF1[jt] = wmma16(a, b, aF1[jt]);
          b = load_b_frag(Wf2t, U_, n0, k0, lane); aF2[jt] = wmma16(a, b, aF2[jt]);
          b = load_b_frag(Wtat, U_, n0, k0, lane); aTA[jt] = wmma16(a, b, aTA[jt]);
          b = load_b_frag(Wtbt, U_, n0, k0, lane); aTB[jt] = wmma16(a, b, aTB[jt]);
        }
      }
#pragma unroll
      for (int jt = 0; jt < 2; ++jt) {
        const int col = j0 + jt * 16 + lcol;
        const float vb1 = bf1[col], vb2 = bf2[col], vba = bta[col], vbb = btb[col];
#pragma unroll
        for (int r = 0; r < 8; ++r) {
          const int m = rowbase + r;
          float ff1 = fast_tanh(aF1[jt][r] + vb1);
          float ff2 = fast_tanh(aF2[jt][r] + vb2);
          float sg = fast_sigmoid((aTA[jt][r] + vba) * t_s[m] + (aTB[jt][r] + vbb));
          float h = ff1 + sg * (ff2 - ff1);
          z_s[m * ZSTR + F_ + col] = (_Float16)h;
        }
      }
    }
    __syncthreads();

    // ---- S4: out[b, step, :] = h_new @ Wout + bout  [16x256]x[256x128] ----
    {
      v8f acc = zero8();
      const int n0 = wave * 16; // 8 waves x 16 cols = 128
      for (int k0 = 0; k0 < H_; k0 += 32) {
        v16h a = load_a_frag(z_s, ZSTR, F_ + k0, lane);
        v16h b = load_b_frag(Wot, H_, n0, k0, lane);
        acc = wmma16(a, b, acc);
      }
      const int col = n0 + lcol;
      const float bias = bo[col];
#pragma unroll
      for (int r = 0; r < 8; ++r) {
        const int m = rowbase + r;
        out[((size_t)(mb * 16 + m) * T_ + step) * OUT_ + col] = acc[r] + bias;
      }
    }
    // next write to h-columns happens after 3 more barriers; x/t staging is disjoint,
    // so no barrier needed here.
  }
}

extern "C" void kernel_launch(void* const* d_in, const int* in_sizes, int n_in,
                              void* d_out, int out_size, void* d_ws, size_t ws_size,
                              hipStream_t stream) {
  (void)in_sizes; (void)n_in; (void)out_size; (void)ws_size;

  const float* x    = (const float*)d_in[0];
  const float* ts   = (const float*)d_in[1];
  const float* W1   = (const float*)d_in[2];
  const float* b1   = (const float*)d_in[3];
  const float* W2   = (const float*)d_in[4];
  const float* b2   = (const float*)d_in[5];
  const float* Wff1 = (const float*)d_in[6];
  const float* bff1 = (const float*)d_in[7];
  const float* Wff2 = (const float*)d_in[8];
  const float* bff2 = (const float*)d_in[9];
  const float* Wta  = (const float*)d_in[10];
  const float* bta  = (const float*)d_in[11];
  const float* Wtb  = (const float*)d_in[12];
  const float* btb  = (const float*)d_in[13];
  const float* Wout = (const float*)d_in[14];
  const float* bout = (const float*)d_in[15];
  float* out = (float*)d_out;

  // f16 transposed weights in workspace (all offsets 32B-aligned)
  char* ws = (char*)d_ws;
  _Float16* W1t  = (_Float16*)(ws + 0);        // [512][384]  393216 B
  _Float16* W2t  = (_Float16*)(ws + 393216);   // [512][512]  524288 B
  _Float16* Wf1t = (_Float16*)(ws + 917504);   // [256][512]  262144 B
  _Float16* Wf2t = (_Float16*)(ws + 1179648);  // [256][512]
  _Float16* Wtat = (_Float16*)(ws + 1441792);  // [256][512]
  _Float16* Wtbt = (_Float16*)(ws + 1703936);  // [256][512]
  _Float16* Wot  = (_Float16*)(ws + 1966080);  // [128][256]   65536 B

  const int TB = 256;
  cfc_w_transpose<<<(FH_ * U_ + TB - 1) / TB, TB, 0, stream>>>(W1, W1t, FH_, U_);
  cfc_w_transpose<<<(U_ * U_ + TB - 1) / TB, TB, 0, stream>>>(W2, W2t, U_, U_);
  cfc_w_transpose<<<(U_ * H_ + TB - 1) / TB, TB, 0, stream>>>(Wff1, Wf1t, U_, H_);
  cfc_w_transpose<<<(U_ * H_ + TB - 1) / TB, TB, 0, stream>>>(Wff2, Wf2t, U_, H_);
  cfc_w_transpose<<<(U_ * H_ + TB - 1) / TB, TB, 0, stream>>>(Wta, Wtat, U_, H_);
  cfc_w_transpose<<<(U_ * H_ + TB - 1) / TB, TB, 0, stream>>>(Wtb, Wtbt, U_, H_);
  cfc_w_transpose<<<(H_ * OUT_ + TB - 1) / TB, TB, 0, stream>>>(Wout, Wot, H_, OUT_);

  cfc_main<<<B_ / 16, 256, 0, stream>>>(x, ts, W1t, b1, W2t, b2, Wf1t, bff1, Wf2t, bff2,
                                        Wtat, bta, Wtbt, btb, Wot, bout, out);
}
